// GlobalCausalIntegrator_52510270161045
// MI455X (gfx1250) — compile-verified, matrix operand order checked by reading, not run on hardware
//
#include <hip/hip_runtime.h>

// ---------------- problem constants ----------------
#define NB    4
#define NT    4
#define CC    768
#define FF    3072
#define NH    12
#define DHD   64
#define SP    256              // H*W
#define LQ    1024             // NT*SP
#define NROW  4096             // NB*LQ

typedef __attribute__((ext_vector_type(16))) _Float16 v16h;
typedef __attribute__((ext_vector_type(8)))  _Float16 v8h;
typedef __attribute__((ext_vector_type(2)))  _Float16 v2h;
typedef __attribute__((ext_vector_type(8)))  float    v8f;

// D = A(16x32 f16) x B(32x16 f16) + C(16x16 f32)
__device__ __forceinline__ v8f wmma_f16(v16h a, v16h b, v8f c) {
  return __builtin_amdgcn_wmma_f32_16x16x32_f16(false, a, false, b, (short)0, c,
                                                false, false);
}

// CDNA5 async global->LDS copy, 16B per lane, tracked by ASYNCcnt.
// ldsOff = LDS byte offset (low 32 bits of flat shared address), g = global addr.
__device__ __forceinline__ void async_b128(unsigned ldsOff, const void* g) {
  asm volatile("global_load_async_to_lds_b128 %0, %1, off"
               :: "v"(ldsOff), "v"((unsigned long long)(size_t)g)
               : "memory");
}
__device__ __forceinline__ void wait_async0() {
#if __has_builtin(__builtin_amdgcn_s_wait_asynccnt)
  __builtin_amdgcn_s_wait_asynccnt(0);
#else
  asm volatile("s_wait_asynccnt 0x0" ::: "memory");
#endif
}

// Load a 16x32 f16 fragment from row-major storage (A-frag rows of A, B-frag
// rows of Bt).  Per ISA 16-bit A layout:
//   lane<16 : row=lane,    K = k0+{0..7} and k0+{16..23}
//   lane>=16: row=lane-16, K = k0+{8..15} and k0+{24..31}
__device__ __forceinline__ v16h frag_load(const _Float16* p, int stride, int lane) {
  const int row = lane & 15;
  const int k0  = (lane >> 4) << 3;
  const _Float16* q = p + (size_t)row * stride + k0;
  v8h lo = *(const v8h*)q;
  v8h hi = *(const v8h*)(q + 16);
  return __builtin_shufflevector(lo, hi, 0,1,2,3,4,5,6,7,8,9,10,11,12,13,14,15);
}

// ---------------- input permute: (B,T,C,H,W) -> tok[NROW,CC] f32+f16 ----------
__global__ __launch_bounds__(256) void permute_in(const float* __restrict__ x,
                                                  float* __restrict__ tokF,
                                                  _Float16* __restrict__ tokH) {
  int idx = blockIdx.x * 256 + threadIdx.x;      // over B*T*C*S = 3145728
  int hw = idx & (SP - 1);
  int i  = idx >> 8;
  int c  = i % CC; i /= CC;
  int t  = i & 3;  int b = i >> 2;
  int row = b * LQ + t * SP + hw;
  float v = x[idx];
  tokF[(size_t)row * CC + c] = v;
  tokH[(size_t)row * CC + c] = (_Float16)v;
}

// ---------------- weight transpose+convert: W[K,N] f32 -> Wt[N,K] f16 ---------
__global__ __launch_bounds__(256) void transpose_w(const float* __restrict__ w,
                                                   _Float16* __restrict__ wt,
                                                   int Kd, int Nd) {
  int idx = blockIdx.x * 256 + threadIdx.x;      // over Kd*Nd, dst-major
  int n = idx / Kd, k = idx - n * Kd;
  wt[idx] = (_Float16)w[(size_t)k * Nd + n];
}

// ---------------- tiled WMMA GEMM: C[M,N] = A[M,K] * Bt[N,K]^T ----------------
// Double-buffered: async global->LDS loads for tile kt+32 overlap WMMAs on kt.
// MODE 0: outF = acc + bias (+ resid)               (f32)
// MODE 1: outH = relu(acc + bias)                   (f16)
// MODE 2: RoPE epilogue -> outH = Q/K [B,NH,L,DH]   (f16)
// MODE 3: transpose     -> outH = Vt [B,NH,DH,L]    (f16)
template <int MODE>
__global__ __launch_bounds__(256) void gemm_wmma(
    const _Float16* __restrict__ A, const _Float16* __restrict__ Bt,
    const float* __restrict__ bias, const float* __restrict__ resid,
    float* __restrict__ outF, _Float16* __restrict__ outH,
    int M, int N, int K) {
  __shared__ __align__(16) _Float16 sA[2][128 * 40];  // stride 40 halves (bank pad)
  __shared__ __align__(16) _Float16 sB[2][128 * 40];
  const int tid  = threadIdx.x;
  const int lane = tid & 31;
  const int wave = tid >> 5;
  const int wm = wave & 3, wn = wave >> 2;            // 4x2 waves -> 128x128 tile
  const int RM = blockIdx.y * 128;
  const int RN = blockIdx.x * 128;

  const unsigned aOff[2] = {(unsigned)(size_t)&sA[0][0], (unsigned)(size_t)&sA[1][0]};
  const unsigned bOff[2] = {(unsigned)(size_t)&sB[0][0], (unsigned)(size_t)&sB[1][0]};

  auto issue = [&](int kt, int buf) {
    for (int c = tid; c < 512; c += 256) {            // 512 chunks of 8 halves
      int r  = c >> 2;
      int k8 = (c & 3) << 3;
      unsigned o = (unsigned)((r * 40 + k8) * 2);
      async_b128(aOff[buf] + o, &A [(size_t)(RM + r) * K + kt + k8]);
      async_b128(bOff[buf] + o, &Bt[(size_t)(RN + r) * K + kt + k8]);
    }
  };

  v8f zero = {};
  v8f acc[2][4];
  for (int i = 0; i < 2; i++)
    for (int j = 0; j < 4; j++) acc[i][j] = zero;

  issue(0, 0);
  int buf = 0;
  for (int kt = 0; kt < K; kt += 32) {
    wait_async0();                                    // my async ops for buf done
    __syncthreads();                                  // => whole tile visible
    if (kt + 32 < K) issue(kt + 32, buf ^ 1);         // prefetch next tile
    const _Float16* cA = sA[buf];
    const _Float16* cB = sB[buf];
    v16h af[2], bf[4];
#pragma unroll
    for (int i = 0; i < 2; i++) af[i] = frag_load(&cA[(wm * 32 + i * 16) * 40], 40, lane);
#pragma unroll
    for (int j = 0; j < 4; j++) bf[j] = frag_load(&cB[(wn * 64 + j * 16) * 40], 40, lane);
#pragma unroll
    for (int i = 0; i < 2; i++)
#pragma unroll
      for (int j = 0; j < 4; j++) acc[i][j] = wmma_f16(af[i], bf[j], acc[i][j]);
    __syncthreads();                                  // done reading buf before refill
    buf ^= 1;
  }

  const int roff = (lane >> 4) << 3;                  // C-layout: lanes>=16 own rows+8
  const int cl   = lane & 15;

  if (MODE == 0) {
    for (int i = 0; i < 2; i++)
      for (int j = 0; j < 4; j++) {
        int n = RN + wn * 64 + j * 16 + cl;
        float bv = bias[n];
#pragma unroll
        for (int r = 0; r < 8; r++) {
          int row = RM + wm * 32 + i * 16 + roff + r;
          float v = acc[i][j][r] + bv;
          if (resid) v += resid[(size_t)row * N + n];
          outF[(size_t)row * N + n] = v;
        }
      }
  } else if (MODE == 1) {
    for (int i = 0; i < 2; i++)
      for (int j = 0; j < 4; j++) {
        int n = RN + wn * 64 + j * 16 + cl;
        float bv = bias[n];
#pragma unroll
        for (int r = 0; r < 8; r++) {
          int row = RM + wm * 32 + i * 16 + roff + r;
          outH[(size_t)row * N + n] = (_Float16)fmaxf(acc[i][j][r] + bv, 0.0f);
        }
      }
  } else if (MODE == 2) {
    // wave owns a head-aligned 64-wide span: rotation pairs (j, j+32) live in
    // acc tiles (i,j) and (i,j+2) at identical lane/VGPR -> fully local RoPE.
    const int head = (RN + wn * 64) >> 6;
    for (int i = 0; i < 2; i++)
      for (int j = 0; j < 2; j++) {
        int jj = j * 16 + cl;                         // 0..31 (half-dim index)
        int nlo = RN + wn * 64 + jj;
        float b1v = bias[nlo], b2v = bias[nlo + 32];
        float invf = exp2f(-(float)jj * 0.4152410118609203f);  // log2(1e4)/32
#pragma unroll
        for (int r = 0; r < 8; r++) {
          int row = RM + wm * 32 + i * 16 + roff + r;
          int b = row >> 10, l = row & (LQ - 1);
          float ang = (float)l * invf;
          float cs = __cosf(ang), sn = __sinf(ang);
          float x1 = acc[i][j][r] + b1v;
          float x2 = acc[i][j + 2][r] + b2v;
          size_t base = (((size_t)b * NH + head) * LQ + l) * DHD;
          outH[base + jj]      = (_Float16)(x1 * cs - x2 * sn);
          outH[base + jj + 32] = (_Float16)(x2 * cs + x1 * sn);
        }
      }
  } else {  // MODE 3: Vt[b,h,dh,l]
    for (int i = 0; i < 2; i++)
      for (int j = 0; j < 4; j++) {
        int n = RN + wn * 64 + j * 16 + cl;
        int head = n >> 6, dh = n & 63;
        float bv = bias[n];
#pragma unroll
        for (int r = 0; r < 8; r++) {
          int row = RM + wm * 32 + i * 16 + roff + r;
          int b = row >> 10, l = row & (LQ - 1);
          outH[(((size_t)b * NH + head) * DHD + dh) * LQ + l] =
              (_Float16)(acc[i][j][r] + bv);
        }
      }
  }
}

// ---------------- flash attention (block-causal), one wave per 16 queries -----
__global__ __launch_bounds__(256) void attn_wmma(
    const _Float16* __restrict__ Q,    // [B,NH,L,DH]
    const _Float16* __restrict__ Km,   // [B,NH,L,DH]
    const _Float16* __restrict__ Vt,   // [B,NH,DH,L]
    _Float16* __restrict__ ctx) {      // [B*L, CC]
  const int lane = threadIdx.x & 31;
  const int wave = threadIdx.x >> 5;
  const int bh = blockIdx.y;
  const int b = bh / NH, h = bh - b * NH;
  const int q0 = (blockIdx.x * 8 + wave) << 4;
  const int kmax = ((q0 >> 8) + 1) << 8;             // block-causal: keys < (t+1)*256
  const size_t qkB = (size_t)bh * LQ * DHD;
  const size_t vB  = (size_t)bh * DHD * LQ;

  // Q fragments (B-operand of S^T = K * Q^T), resident for the whole sweep
  v16h qf0 = frag_load(Q + qkB + (size_t)q0 * DHD,      DHD, lane);
  v16h qf1 = frag_load(Q + qkB + (size_t)q0 * DHD + 32, DHD, lane);

  v8f zero = {};
  v8f cacc[4];                                       // ctx^T: 4 dh-tiles x 16 q
  for (int t = 0; t < 4; t++) cacc[t] = zero;
  float m_run = -3.0e38f, l_run = 0.0f;
  const float SCL = 0.125f * 1.44269504088896341f;   // 1/sqrt(64) * log2(e)

  for (int kb = 0; kb < kmax; kb += 32) {
    if (kb + 32 < kmax) {                            // prefetch next key tile (L2 walk)
      const _Float16* nk =
          Km + qkB + (size_t)(kb + 32 + (lane & 15)) * DHD + ((lane >> 4) << 3);
      __builtin_prefetch(nk, 0, 3);
      __builtin_prefetch(nk + 16 * DHD, 0, 3);
    }
    v16h a00 = frag_load(Km + qkB + (size_t)kb * DHD,             DHD, lane);
    v16h a01 = frag_load(Km + qkB + (size_t)kb * DHD + 32,        DHD, lane);
    v16h a10 = frag_load(Km + qkB + (size_t)(kb + 16) * DHD,      DHD, lane);
    v16h a11 = frag_load(Km + qkB + (size_t)(kb + 16) * DHD + 32, DHD, lane);
    v8f s0 = wmma_f16(a00, qf0, zero); s0 = wmma_f16(a01, qf1, s0);
    v8f s1 = wmma_f16(a10, qf0, zero); s1 = wmma_f16(a11, qf1, s1);

    float mloc = -3.0e38f;
#pragma unroll
    for (int r = 0; r < 8; r++) {
      s0[r] *= SCL; s1[r] *= SCL;
      mloc = fmaxf(mloc, fmaxf(s0[r], s1[r]));
    }
    mloc = fmaxf(mloc, __shfl_xor(mloc, 16, 32));    // combine key halves
    float m_new = fmaxf(m_run, mloc);
    float alpha = exp2f(m_run - m_new);
    float p0[8], p1[8], ls = 0.0f;
#pragma unroll
    for (int r = 0; r < 8; r++) {
      p0[r] = exp2f(s0[r] - m_new);
      p1[r] = exp2f(s1[r] - m_new);
      ls += p0[r] + p1[r];
    }
    ls += __shfl_xor(ls, 16, 32);
    l_run = l_run * alpha + ls;
    m_run = m_new;
#pragma unroll
    for (int t = 0; t < 4; t++)
#pragma unroll
      for (int r = 0; r < 8; r++) cacc[t][r] *= alpha;

    // C-layout half-split (rows+8 in lanes>=16) == B-frag K-split: pack directly
    v16h pb;
#pragma unroll
    for (int r = 0; r < 8; r++) { pb[r] = (_Float16)p0[r]; pb[r + 8] = (_Float16)p1[r]; }
#pragma unroll
    for (int t = 0; t < 4; t++) {
      v16h va = frag_load(Vt + vB + (size_t)(t * 16) * LQ + kb, LQ, lane);
      cacc[t] = wmma_f16(va, pb, cacc[t]);           // ctx^T += V^T * P
    }
  }

  float inv_l = 1.0f / l_run;
  const int q    = q0 + (lane & 15);
  const int roff = (lane >> 4) << 3;
  _Float16* cp = ctx + (size_t)(b * LQ + q) * CC + h * DHD;
#pragma unroll
  for (int t = 0; t < 4; t++)
#pragma unroll
    for (int r = 0; r < 8; r += 2) {
      int dh = t * 16 + roff + r;
      v2h pr = {(_Float16)(cacc[t][r] * inv_l), (_Float16)(cacc[t][r + 1] * inv_l)};
      *(v2h*)(cp + dh) = pr;
    }
}

// ---------------- layernorm (one block per row) -------------------------------
__global__ __launch_bounds__(256) void ln_mid(const float* __restrict__ z,
                                              const float* __restrict__ g,
                                              const float* __restrict__ be,
                                              float* __restrict__ yf,
                                              _Float16* __restrict__ yh) {
  __shared__ float red[256];
  const int row = blockIdx.x, tid = threadIdx.x;
  const float* x = z + (size_t)row * CC;
  float s = 0.0f;
  for (int c = tid; c < CC; c += 256) s += x[c];
  red[tid] = s; __syncthreads();
  for (int st = 128; st > 0; st >>= 1) { if (tid < st) red[tid] += red[tid + st]; __syncthreads(); }
  float mu = red[0] * (1.0f / CC);
  __syncthreads();
  float v = 0.0f;
  for (int c = tid; c < CC; c += 256) { float d = x[c] - mu; v += d * d; }
  red[tid] = v; __syncthreads();
  for (int st = 128; st > 0; st >>= 1) { if (tid < st) red[tid] += red[tid + st]; __syncthreads(); }
  float rstd = rsqrtf(red[0] * (1.0f / CC) + 1e-5f);
  for (int c = tid; c < CC; c += 256) {
    float y = (x[c] - mu) * rstd * g[c] + be[c];
    yf[(size_t)row * CC + c] = y;
    yh[(size_t)row * CC + c] = (_Float16)y;
  }
}

__global__ __launch_bounds__(256) void ln_final(const float* __restrict__ z,
                                                const float* __restrict__ g,
                                                const float* __restrict__ be,
                                                float* __restrict__ out) {
  __shared__ float red[256];
  const int row = blockIdx.x, tid = threadIdx.x;
  const float* x = z + (size_t)row * CC;
  float s = 0.0f;
  for (int c = tid; c < CC; c += 256) s += x[c];
  red[tid] = s; __syncthreads();
  for (int st = 128; st > 0; st >>= 1) { if (tid < st) red[tid] += red[tid + st]; __syncthreads(); }
  float mu = red[0] * (1.0f / CC);
  __syncthreads();
  float v = 0.0f;
  for (int c = tid; c < CC; c += 256) { float d = x[c] - mu; v += d * d; }
  red[tid] = v; __syncthreads();
  for (int st = 128; st > 0; st >>= 1) { if (tid < st) red[tid] += red[tid + st]; __syncthreads(); }
  float rstd = rsqrtf(red[0] * (1.0f / CC) + 1e-5f);
  const int b = row >> 10, l = row & (LQ - 1);
  const int t = l >> 8, hw = l & (SP - 1);
  for (int c = tid; c < CC; c += 256) {
    float y = (x[c] - mu) * rstd * g[c] + be[c];
    out[(((size_t)b * NT + t) * CC + c) * SP + hw] = y;   // (B,T,C,H,W)
  }
}

// ---------------- launch ------------------------------------------------------
extern "C" void kernel_launch(void* const* d_in, const int* in_sizes, int n_in,
                              void* d_out, int out_size, void* d_ws, size_t ws_size,
                              hipStream_t stream) {
  (void)in_sizes; (void)n_in; (void)out_size; (void)ws_size;
  const float* x   = (const float*)d_in[0];
  const float* Wq  = (const float*)d_in[1];  const float* bq  = (const float*)d_in[2];
  const float* Wk  = (const float*)d_in[3];  const float* bk  = (const float*)d_in[4];
  const float* Wv  = (const float*)d_in[5];  const float* bv  = (const float*)d_in[6];
  const float* Wo  = (const float*)d_in[7];  const float* bo  = (const float*)d_in[8];
  const float* W1  = (const float*)d_in[9];  const float* b1  = (const float*)d_in[10];
  const float* W2  = (const float*)d_in[11]; const float* b2  = (const float*)d_in[12];
  const float* g1  = (const float*)d_in[13]; const float* be1 = (const float*)d_in[14];
  const float* g2  = (const float*)d_in[15]; const float* be2 = (const float*)d_in[16];

  char* ws = (char*)d_ws;
  size_t off = 0;
  auto alloc = [&](size_t bytes) -> char* {
    char* p = ws + off;
    off = (off + bytes + 255) & ~(size_t)255;
    return p;
  };
  float*    tokF = (float*)   alloc((size_t)NROW * CC * 4);
  _Float16* tokH = (_Float16*)alloc((size_t)NROW * CC * 2);
  _Float16* WqT  = (_Float16*)alloc((size_t)CC * CC * 2);
  _Float16* WkT  = (_Float16*)alloc((size_t)CC * CC * 2);
  _Float16* WvT  = (_Float16*)alloc((size_t)CC * CC * 2);
  _Float16* WoT  = (_Float16*)alloc((size_t)CC * CC * 2);
  _Float16* W1T  = (_Float16*)alloc((size_t)CC * FF * 2);
  _Float16* W2T  = (_Float16*)alloc((size_t)FF * CC * 2);
  _Float16* ctxH = (_Float16*)alloc((size_t)NROW * CC * 2);
  float*    z1   = (float*)   alloc((size_t)NROW * CC * 4);
  float*    y1F  = (float*)   alloc((size_t)NROW * CC * 4);
  _Float16* y1H  = (_Float16*)alloc((size_t)NROW * CC * 2);
  float*    z2   = (float*)   alloc((size_t)NROW * CC * 4);
  const size_t qkvOne = (size_t)NB * NH * LQ * DHD * 2;    // 6 MiB each
  char* regionA = alloc((size_t)NROW * FF * 2);            // 24 MiB > 3*qkvOne
  _Float16* qH  = (_Float16*)regionA;
  _Float16* kH  = (_Float16*)(regionA + qkvOne);
  _Float16* vtH = (_Float16*)(regionA + 2 * qkvOne);
  _Float16* hH  = (_Float16*)regionA;                      // reuse after attention

  permute_in<<<NROW * CC / 256, 256, 0, stream>>>(x, tokF, tokH);
  transpose_w<<<CC * CC / 256, 256, 0, stream>>>(Wq, WqT, CC, CC);
  transpose_w<<<CC * CC / 256, 256, 0, stream>>>(Wk, WkT, CC, CC);
  transpose_w<<<CC * CC / 256, 256, 0, stream>>>(Wv, WvT, CC, CC);
  transpose_w<<<CC * CC / 256, 256, 0, stream>>>(Wo, WoT, CC, CC);
  transpose_w<<<CC * FF / 256, 256, 0, stream>>>(W1, W1T, CC, FF);
  transpose_w<<<CC * FF / 256, 256, 0, stream>>>(W2, W2T, FF, CC);

  gemm_wmma<2><<<dim3(CC / 128, NROW / 128), 256, 0, stream>>>(tokH, WqT, bq, nullptr, nullptr, qH,  NROW, CC, CC);
  gemm_wmma<2><<<dim3(CC / 128, NROW / 128), 256, 0, stream>>>(tokH, WkT, bk, nullptr, nullptr, kH,  NROW, CC, CC);
  gemm_wmma<3><<<dim3(CC / 128, NROW / 128), 256, 0, stream>>>(tokH, WvT, bv, nullptr, nullptr, vtH, NROW, CC, CC);

  attn_wmma<<<dim3(LQ / 128, NB * NH), 256, 0, stream>>>(qH, kH, vtH, ctxH);

  gemm_wmma<0><<<dim3(CC / 128, NROW / 128), 256, 0, stream>>>(ctxH, WoT, bo, tokF, z1, nullptr, NROW, CC, CC);
  ln_mid<<<NROW, 256, 0, stream>>>(z1, g1, be1, y1F, y1H);
  gemm_wmma<1><<<dim3(FF / 128, NROW / 128), 256, 0, stream>>>(y1H, W1T, b1, nullptr, nullptr, hH, NROW, FF, CC);
  gemm_wmma<0><<<dim3(CC / 128, NROW / 128), 256, 0, stream>>>(hH, W2T, b2, y1F, z2, nullptr, NROW, CC, FF);
  ln_final<<<NROW, 256, 0, stream>>>(z2, g2, be2, (float*)d_out);
}
